// GCN_52226802320176
// MI455X (gfx1250) — compile-verified, matrix operand order checked by reading, not run on hardware
//
#include <hip/hip_runtime.h>

#define N_NODES 50000
#define N_EDGES 800000
#define DIM     128
#define BN_EPS  1e-5f
#define ROWS_PER_BLK 125   // 50000 = 400 * 125

typedef __attribute__((ext_vector_type(16))) __bf16 v16bf;
typedef __attribute__((ext_vector_type(8)))  float  v8f;

__device__ __forceinline__ unsigned short f2bf(float f) {
    // round-to-nearest-even fp32 -> bf16
    unsigned u = __float_as_uint(f);
    unsigned r = 0x7FFFu + ((u >> 16) & 1u);
    return (unsigned short)((u + r) >> 16);
}

// ---------------- utility kernels ----------------

__global__ void k_zero(float* __restrict__ p, int n) {
    int i = blockIdx.x * blockDim.x + threadIdx.x;
    if (i < n) p[i] = 0.f;
}

__global__ void k_deg(const int* __restrict__ src, const int* __restrict__ dst,
                      float* __restrict__ deg_out, float* __restrict__ deg_in) {
    int e = blockIdx.x * blockDim.x + threadIdx.x;
    if (e < N_EDGES) {
        atomicAdd(&deg_out[src[e]], 1.f);
        atomicAdd(&deg_in [dst[e]], 1.f);
    }
}

__global__ void k_rsqrt_clamp(float* __restrict__ p, int n) {
    int i = blockIdx.x * blockDim.x + threadIdx.x;
    if (i < n) p[i] = rsqrtf(fmaxf(p[i], 1.f));
}

// fp32 W[k][n] -> bf16 Wt[n][k]  (transposed so B fragments are contiguous in K)
__global__ void k_cvt_bf16_t(const float* __restrict__ w, unsigned short* __restrict__ wt) {
    int i = blockIdx.x * blockDim.x + threadIdx.x;   // i over DIM*DIM, coalesced read
    if (i < DIM * DIM) {
        int k = i >> 7, n = i & (DIM - 1);
        wt[n * DIM + k] = f2bf(w[i]);
    }
}

// ---------------- WMMA GEMM: out[N,D] = (x * rowscale) @ W ----------------
// block = 256 threads = 8 waves; block handles 16 rows x 128 cols.
// Each wave owns one 16x16 output tile, K-loop of 4 x (16x16x32 bf16 WMMA).
// A fragments: ds_load_b128 from staged scaled-bf16 LDS tile.
// B fragments: global_load_b128 from pre-transposed bf16 weights (L1/L2 resident).

__global__ __launch_bounds__(256)
void k_gemm_bf16(const float* __restrict__ x, const float* __restrict__ rowscale,
                 const unsigned short* __restrict__ Wt, float* __restrict__ out) {
    __shared__ unsigned short sA[16][DIM + 8];   // row stride 272 B: 16B-aligned, conflict-free

    const int rowbase = blockIdx.x * 16;
    const int tid = threadIdx.x;

    // cooperative staging: 16x128 fp32 -> scale -> bf16 -> LDS (float4 loads, 2/thread)
    {
        const int i0 = tid * 2;
#pragma unroll
        for (int j = 0; j < 2; ++j) {
            int i = i0 + j;                 // 0..511 quads
            int r = i >> 5, c = (i & 31) * 4;
            float s = rowscale[rowbase + r];
            float4 v = *(const float4*)(x + (rowbase + r) * DIM + c);
            sA[r][c + 0] = f2bf(v.x * s);
            sA[r][c + 1] = f2bf(v.y * s);
            sA[r][c + 2] = f2bf(v.z * s);
            sA[r][c + 3] = f2bf(v.w * s);
        }
    }
    __syncthreads();

    const int wave = tid >> 5;        // 0..7 -> column tile
    const int lane = tid & 31;
    const int hi   = lane >> 4;       // half-wave: K sub-stripe select / M half
    const int m    = lane & 15;       // A row in tile == B column in tile
    const int gcol = wave * 16 + m;   // global output column

    const uint4* sa = (const uint4*)&sA[m][0];          // 8 bf16 per uint4
    const uint4* wb = (const uint4*)(Wt + gcol * DIM);  // transposed weight row

    union { v16bf v; uint4 q[2]; } a, b;
    v8f acc = {};

#pragma unroll
    for (int kc = 0; kc < 4; ++kc) {
        // fragment element e: k = kc*32 + (e<8 ? hi*8+e : 16+hi*8+(e-8))
        // -> two contiguous 16B chunks at ushort offsets kc*32+hi*8 and kc*32+16+hi*8
        a.q[0] = sa[kc * 4 + hi];
        a.q[1] = sa[kc * 4 + 2 + hi];
        b.q[0] = wb[kc * 4 + hi];
        b.q[1] = wb[kc * 4 + 2 + hi];
        acc = __builtin_amdgcn_wmma_f32_16x16x32_bf16(
            /*neg_a=*/false, a.v, /*neg_b=*/false, b.v,
            /*c_mod=*/(short)0, acc, /*reuse_a=*/false, /*reuse_b=*/false);
    }

#pragma unroll
    for (int r = 0; r < 8; ++r) {
        out[(rowbase + hi * 8 + r) * DIM + gcol] = acc[r];
    }
}

// ---------------- SpMM scatter: agg[dst] += h[src] ----------------
// block = 256 threads = 8 edges; each lane moves 4 contiguous floats (B128 gather).

__global__ __launch_bounds__(256)
void k_spmm(const float* __restrict__ h, const int* __restrict__ src,
            const int* __restrict__ dst, float* __restrict__ agg) {
    const int tid  = threadIdx.x;
    const int e    = blockIdx.x * 8 + (tid >> 5);
    const int lane = tid & 31;
    const int f    = lane * 4;
    const int s = src[e] * DIM + f;
    const int d = dst[e] * DIM + f;
    float4 v = *(const float4*)(h + s);
    atomicAdd(&agg[d + 0], v.x);
    atomicAdd(&agg[d + 1], v.y);
    atomicAdd(&agg[d + 2], v.z);
    atomicAdd(&agg[d + 3], v.w);
}

// ---------------- fused: x = agg * din_isqrt[row] + b[col]; accumulate BN stats ----

__global__ __launch_bounds__(128)
void k_scale_bias_stats(float* __restrict__ x, const float* __restrict__ din_isqrt,
                        const float* __restrict__ bias,
                        float* __restrict__ ssum, float* __restrict__ ssq) {
    const int col  = threadIdx.x;        // one column per thread, coalesced rows
    const int row0 = blockIdx.x * ROWS_PER_BLK;
    const float b = bias[col];
    float s = 0.f, q = 0.f;
    for (int i = 0; i < ROWS_PER_BLK; ++i) {
        const int row = row0 + i;
        float v = x[row * DIM + col] * din_isqrt[row] + b;
        x[row * DIM + col] = v;
        s += v;
        q += v * v;
    }
    atomicAdd(&ssum[col], s);
    atomicAdd(&ssq [col], q);
}

// ---------------- fused BatchNorm(normalize) + PReLU, in place ----------------

__global__ __launch_bounds__(256)
void k_bn_prelu(float* __restrict__ x,
                const float* __restrict__ ssum, const float* __restrict__ ssq,
                const float* __restrict__ gamma, const float* __restrict__ beta,
                const float* __restrict__ a) {
    const int i   = blockIdx.x * 256 + threadIdx.x;
    const int col = i & (DIM - 1);
    const float invN = 1.f / (float)N_NODES;
    float mean = ssum[col] * invN;
    float var  = ssq [col] * invN - mean * mean;
    float inv  = rsqrtf(var + BN_EPS);
    float v = gamma[col] * (x[i] - mean) * inv + beta[col];
    float al = a[0];
    x[i] = (v >= 0.f) ? v : al * v;
}

// ---------------- host-side orchestration ----------------

extern "C" void kernel_launch(void* const* d_in, const int* in_sizes, int n_in,
                              void* d_out, int out_size, void* d_ws, size_t ws_size,
                              hipStream_t stream) {
    (void)in_sizes; (void)n_in; (void)out_size; (void)ws_size;

    const float* feat   = (const float*)d_in[0];
    const int*   src    = (const int*)  d_in[1];
    const int*   dst    = (const int*)  d_in[2];
    const float* W1     = (const float*)d_in[3];
    const float* b1     = (const float*)d_in[4];
    const float* gamma1 = (const float*)d_in[5];
    const float* beta1  = (const float*)d_in[6];
    const float* a1     = (const float*)d_in[7];
    const float* W2     = (const float*)d_in[8];
    const float* b2     = (const float*)d_in[9];
    const float* gamma2 = (const float*)d_in[10];
    const float* beta2  = (const float*)d_in[11];
    const float* a2     = (const float*)d_in[12];

    float* out = (float*)d_out;

    // workspace layout (Wt pointers are 16B-aligned: float offset total is even *4B)
    float* ws       = (float*)d_ws;
    float* deg_out  = ws;                                  // N (becomes dout^-1/2)
    float* deg_in   = deg_out + N_NODES;                   // N (becomes din^-1/2)
    float* h        = deg_in  + N_NODES;                   // N*D
    float* xbuf     = h + (size_t)N_NODES * DIM;           // N*D
    float* ssum     = xbuf + (size_t)N_NODES * DIM;        // D
    float* ssq      = ssum + DIM;                          // D (contiguous with ssum)
    unsigned short* Wt1 = (unsigned short*)(ssq + DIM);    // D*D bf16, transposed
    unsigned short* Wt2 = Wt1 + DIM * DIM;                 // D*D bf16, transposed

    const int ND = N_NODES * DIM;

    // --- degrees & normalization factors ---
    k_zero       <<<(2 * N_NODES + 255) / 256, 256, 0, stream>>>(deg_out, 2 * N_NODES);
    k_deg        <<<(N_EDGES + 255) / 256,     256, 0, stream>>>(src, dst, deg_out, deg_in);
    k_rsqrt_clamp<<<(2 * N_NODES + 255) / 256, 256, 0, stream>>>(deg_out, 2 * N_NODES);

    // --- weights to transposed bf16 (L2-resident) ---
    k_cvt_bf16_t<<<(DIM * DIM + 255) / 256, 256, 0, stream>>>(W1, Wt1);
    k_cvt_bf16_t<<<(DIM * DIM + 255) / 256, 256, 0, stream>>>(W2, Wt2);

    // --- layer 1 ---
    k_gemm_bf16<<<N_NODES / 16, 256, 0, stream>>>(feat, deg_out, Wt1, h);
    k_zero     <<<(ND + 255) / 256, 256, 0, stream>>>(xbuf, ND);
    k_spmm     <<<N_EDGES / 8, 256, 0, stream>>>(h, src, dst, xbuf);
    k_zero     <<<1, 256, 0, stream>>>(ssum, 2 * DIM);
    k_scale_bias_stats<<<N_NODES / ROWS_PER_BLK, 128, 0, stream>>>(xbuf, deg_in, b1, ssum, ssq);
    k_bn_prelu <<<ND / 256, 256, 0, stream>>>(xbuf, ssum, ssq, gamma1, beta1, a1);

    // --- layer 2 (scatter straight into d_out) ---
    k_gemm_bf16<<<N_NODES / 16, 256, 0, stream>>>(xbuf, deg_out, Wt2, h);
    k_zero     <<<(ND + 255) / 256, 256, 0, stream>>>(out, ND);
    k_spmm     <<<N_EDGES / 8, 256, 0, stream>>>(h, src, dst, out);
    k_zero     <<<1, 256, 0, stream>>>(ssum, 2 * DIM);
    k_scale_bias_stats<<<N_NODES / ROWS_PER_BLK, 128, 0, stream>>>(out, deg_in, b2, ssum, ssq);
    k_bn_prelu <<<ND / 256, 256, 0, stream>>>(out, ssum, ssq, gamma2, beta2, a2);
}
